// MemoryCausalSelfAttention_81879256531230
// MI455X (gfx1250) — matrix operand order repaired and checked by reading, not compiled, not run on hardware
//
#include <hip/hip_runtime.h>

typedef __attribute__((ext_vector_type(16))) _Float16 v16h;
typedef __attribute__((ext_vector_type(8)))  _Float16 v8h;
typedef __attribute__((ext_vector_type(8)))  float    v8f;

// Problem constants: B=4, T=1024, C=1024, NH=16, HD=64
// T_mem = 129+257+257+1024 = 1667
#define TDIM   1024
#define CDIM   1024
#define NHEAD  16
#define HDIM   64
#define TMEM   1667
#define TMEMP  1680   // padded stride for transposed V (multiple of 16)
#define K3OFF  386    // 129 + 257
#define KNEW   643    // 129 + 257 + 257

// ---------------------------------------------------------------------------
// Kernel 0: bulk f32 -> f16 conversion (x and both weight matrices, once).
// ---------------------------------------------------------------------------
__global__ void k_cvt(const float* __restrict__ src, _Float16* __restrict__ dst, long n)
{
    long i = ((long)blockIdx.x * blockDim.x + threadIdx.x) * 8;
    if (i >= n) return;
    v8f f = *(const v8f*)(src + i);
    v8h h;
#pragma unroll
    for (int j = 0; j < 8; ++j) h[j] = (_Float16)f[j];
    *(v8h*)(dst + i) = h;
}

// ---------------------------------------------------------------------------
// GEMM: Out[M,N] = A[M,K](f16) @ W[N,K](f16)^T + bias.  One wave per 64x64
// tile (16 accumulators, ~halves L2 weight traffic vs 32-row tiles),
// software pipelined: next k-step fragments load while current 16 WMMAs issue.
// ---------------------------------------------------------------------------
template <int N, bool OUT_F32>
__global__ __launch_bounds__(32) void k_gemm(const _Float16* __restrict__ A,
                                             const _Float16* __restrict__ W,
                                             const float* __restrict__ bias,
                                             void* __restrict__ OutP)
{
    constexpr int K = 1024;
    const int nt64 = N / 64;
    int wid = blockIdx.x;
    int m0 = (wid / nt64) * 64;
    int n0 = (wid % nt64) * 64;
    int L = threadIdx.x & 31, hf = L >> 4, lm = L & 15;

    const _Float16* ar[4];
#pragma unroll
    for (int mi = 0; mi < 4; ++mi)
        ar[mi] = A + (size_t)(m0 + mi * 16 + lm) * K;
    const _Float16* br[4];
#pragma unroll
    for (int t = 0; t < 4; ++t)
        br[t] = W + (size_t)(n0 + t * 16 + lm) * K + hf * 16;

    auto loadA = [&](const _Float16* p, int k0) -> v16h {
        v8h x0 = *(const v8h*)(p + k0 + hf * 16);
        v8h x1 = *(const v8h*)(p + k0 + hf * 16 + 8);
        return __builtin_shufflevector(x0, x1, 0, 1, 2, 3, 4, 5, 6, 7,
                                       8, 9, 10, 11, 12, 13, 14, 15);
    };

    v8f acc[16] = {};
    v16h a[4], b[4];
#pragma unroll
    for (int mi = 0; mi < 4; ++mi) a[mi] = loadA(ar[mi], 0);
#pragma unroll
    for (int t = 0; t < 4; ++t) b[t] = *(const v16h*)(br[t]);

    for (int k0 = 32; k0 < K; k0 += 32) {
        // issue next-step loads first (pipelined), then compute current step
        v16h na[4], nb[4];
#pragma unroll
        for (int mi = 0; mi < 4; ++mi) na[mi] = loadA(ar[mi], k0);
#pragma unroll
        for (int t = 0; t < 4; ++t) nb[t] = *(const v16h*)(br[t] + k0);
        __builtin_prefetch(ar[0] + k0 + 256, 0, 1);
        __builtin_prefetch(br[0] + k0 + 256, 0, 1);
#pragma unroll
        for (int mi = 0; mi < 4; ++mi)
#pragma unroll
            for (int t = 0; t < 4; ++t)
                acc[mi * 4 + t] = __builtin_amdgcn_wmma_f32_16x16x32_f16(
                    false, a[mi], false, b[t], (short)0, acc[mi * 4 + t], false, false);
#pragma unroll
        for (int mi = 0; mi < 4; ++mi) a[mi] = na[mi];
#pragma unroll
        for (int t = 0; t < 4; ++t) b[t] = nb[t];
    }
#pragma unroll
    for (int mi = 0; mi < 4; ++mi)
#pragma unroll
        for (int t = 0; t < 4; ++t)
            acc[mi * 4 + t] = __builtin_amdgcn_wmma_f32_16x16x32_f16(
                false, a[mi], false, b[t], (short)0, acc[mi * 4 + t], false, false);

#pragma unroll
    for (int mi = 0; mi < 4; ++mi)
#pragma unroll
        for (int t = 0; t < 4; ++t) {
            int col = n0 + t * 16 + lm;
            float bc = bias[col];
#pragma unroll
            for (int r = 0; r < 8; ++r) {
                int row = m0 + mi * 16 + hf * 8 + r;
                float v = acc[mi * 4 + t][r] + bc;
                if constexpr (OUT_F32) ((float*)OutP)[(size_t)row * N + col] = v;
                else ((_Float16*)OutP)[(size_t)row * N + col] = (_Float16)v;
            }
        }
}

// ---------------------------------------------------------------------------
// Kernel 2: compress mem_k3 / mem_v3 (depthwise conv k=2,s=2,p=1 + LayerNorm)
// ---------------------------------------------------------------------------
__global__ __launch_bounds__(32) void k_compress(
    const float* __restrict__ mk3, const float* __restrict__ mv3,
    const float* __restrict__ wk,  const float* __restrict__ gk, const float* __restrict__ bk,
    const float* __restrict__ wv,  const float* __restrict__ gv, const float* __restrict__ bv,
    _Float16* __restrict__ allk, _Float16* __restrict__ allvT)
{
    constexpr int LIN = 513, LOUT = 257;
    int isV = blockIdx.y;
    int bh = blockIdx.x / LOUT;
    int l  = blockIdx.x % LOUT;
    const float* src = isV ? mv3 : mk3;
    const float* w   = isV ? wv  : wk;
    const float* g   = isV ? gv  : gk;
    const float* bb  = isV ? bv  : bk;
    int lane = threadIdx.x & 31;

    const float* base = src + (size_t)bh * LIN * HDIM;
    float y[2];
#pragma unroll
    for (int e = 0; e < 2; ++e) {
        int d = lane + e * 32;
        float acc = w[d * 2 + 1] * base[(size_t)(2 * l) * HDIM + d];
        if (l > 0) acc += w[d * 2 + 0] * base[(size_t)(2 * l - 1) * HDIM + d];
        y[e] = acc;
    }
    float s  = y[0] + y[1];
    float sq = y[0] * y[0] + y[1] * y[1];
#pragma unroll
    for (int m = 1; m < 32; m <<= 1) { s += __shfl_xor(s, m); sq += __shfl_xor(sq, m); }
    float mean = s * (1.0f / 64.0f);
    float var  = sq * (1.0f / 64.0f) - mean * mean;
    float rstd = rsqrtf(var + 1e-5f);
#pragma unroll
    for (int e = 0; e < 2; ++e) {
        int d = lane + e * 32;
        float o = (y[e] - mean) * rstd * g[d] + bb[d];
        if (isV) allvT[(size_t)(bh * HDIM + d) * TMEMP + (K3OFF + l)] = (_Float16)o;
        else     allk[((size_t)bh * TMEM + (K3OFF + l)) * HDIM + d]   = (_Float16)o;
    }
}

// ---------------------------------------------------------------------------
// Kernel 3: pack mem1/mem2 (f32) + k_new/v_new (f16 from qkv) into contiguous
// all_k [bh,TMEM,64] and transposed all_v^T [bh,64,TMEMP].
// ---------------------------------------------------------------------------
__global__ void k_pack(const float* __restrict__ mk1, const float* __restrict__ mv1,
                       const float* __restrict__ mk2, const float* __restrict__ mv2,
                       const _Float16* __restrict__ qkv,
                       _Float16* __restrict__ allk, _Float16* __restrict__ allvT)
{
    constexpr int ROWS = 1410; // 129 + 257 + 1024
    long total = (long)64 * ROWS * HDIM;
    long tid = (long)blockIdx.x * blockDim.x + threadIdx.x;
    if (tid >= total) return;
    int  d    = (int)(tid & 63);
    long rest = tid >> 6;
    int  jj   = (int)(rest % ROWS);
    int  bh   = (int)(rest / ROWS);
    int  b    = bh >> 4, h = bh & 15;
    float kv, vv;
    int j;
    if (jj < 129) {
        j = jj;
        size_t o = ((size_t)bh * 129 + jj) * HDIM + d;
        kv = mk1[o]; vv = mv1[o];
    } else if (jj < K3OFF) {
        j = jj;
        size_t o = ((size_t)bh * 257 + (jj - 129)) * HDIM + d;
        kv = mk2[o]; vv = mv2[o];
    } else {
        int t = jj - K3OFF;            // 0..1023
        j = KNEW + t;
        size_t base = ((size_t)(b * TDIM + t)) * (3 * CDIM) + h * HDIM + d;
        kv = (float)qkv[base + CDIM];
        vv = (float)qkv[base + 2 * CDIM];
    }
    allk[((size_t)bh * TMEM + j) * HDIM + d]   = (_Float16)kv;
    allvT[((size_t)bh * HDIM + d) * TMEMP + j] = (_Float16)vv;
}

// ---------------------------------------------------------------------------
// Kernel 4: flash-style causal attention, 32 keys per block iteration.
// S: 2 tiles x (2 WMMA, K=64); one softmax reduction pass per 32 keys;
// P relaid D->A via LDS into a FULL K=32 fragment; PV: 4 WMMAs, no idle lanes.
// ---------------------------------------------------------------------------
__global__ __launch_bounds__(32) void k_attn(const _Float16* __restrict__ qkv,
                                             const _Float16* __restrict__ allk,
                                             const _Float16* __restrict__ allvT,
                                             _Float16* __restrict__ yatt)
{
    __shared__ _Float16 plds[512]; // 16 rows x 32 keys
    int gid = blockIdx.x;
    int qt = gid & 63;
    int bh = gid >> 6;
    int b = bh >> 4, h = bh & 15;
    int q0 = qt * 16;
    int L = threadIdx.x & 31, hf = L >> 4, lm = L & 15;
    const float scale = 0.125f; // 1/sqrt(64)

    // Q fragments (A layout), pre-scaled
    const _Float16* qrow = qkv + ((size_t)(b * TDIM + q0 + lm)) * (3 * CDIM) + h * HDIM;
    v16h aq[2];
#pragma unroll
    for (int f = 0; f < 2; ++f) {
        v8h h0 = *(const v8h*)(qrow + f * 32 + hf * 16);
        v8h h1 = *(const v8h*)(qrow + f * 32 + hf * 16 + 8);
#pragma unroll
        for (int i = 0; i < 8; ++i) {
            aq[f][i]     = (_Float16)((float)h0[i] * scale);
            aq[f][8 + i] = (_Float16)((float)h1[i] * scale);
        }
    }

    v8f o[4] = {};
    float mr[8], lr[8];
#pragma unroll
    for (int r = 0; r < 8; ++r) { mr[r] = -3.0e38f; lr[r] = 0.0f; }

    const _Float16* kb = allk + (size_t)bh * TMEM * HDIM;
    const _Float16* vb = allvT + (size_t)bh * HDIM * TMEMP;

    for (int j0 = 0; j0 <= q0; j0 += 32) {
        // ---- S = (scale*Q) @ K^T for 32 keys (2 tiles of 16) ----
        const _Float16* kr0 = kb + (size_t)(j0 + lm) * HDIM + hf * 16;
        const _Float16* kr1 = kr0 + (size_t)16 * HDIM;
        v16h bk00 = *(const v16h*)(kr0);
        v16h bk01 = *(const v16h*)(kr0 + 32);
        v16h bk10 = *(const v16h*)(kr1);
        v16h bk11 = *(const v16h*)(kr1 + 32);
        v8f s0 = {}, s1 = {};
        s0 = __builtin_amdgcn_wmma_f32_16x16x32_f16(false, aq[0], false, bk00, (short)0, s0, false, false);
        s0 = __builtin_amdgcn_wmma_f32_16x16x32_f16(false, aq[1], false, bk01, (short)0, s0, false, false);
        s1 = __builtin_amdgcn_wmma_f32_16x16x32_f16(false, aq[0], false, bk10, (short)0, s1, false, false);
        s1 = __builtin_amdgcn_wmma_f32_16x16x32_f16(false, aq[1], false, bk11, (short)0, s1, false, false);

        // ---- online softmax across both tiles ----
        float alpha[8];
#pragma unroll
        for (int r = 0; r < 8; ++r) {
            int rowg = q0 + hf * 8 + r;
            float sv0 = (j0 + lm      <= rowg) ? s0[r] : -3.0e38f;
            float sv1 = (j0 + 16 + lm <= rowg) ? s1[r] : -3.0e38f;
            float red = fmaxf(sv0, sv1);
            red = fmaxf(red, __shfl_xor(red, 1));
            red = fmaxf(red, __shfl_xor(red, 2));
            red = fmaxf(red, __shfl_xor(red, 4));
            red = fmaxf(red, __shfl_xor(red, 8));
            float mnew = fmaxf(mr[r], red);
            float a_ = __expf(mr[r] - mnew);
            float p0 = __expf(sv0 - mnew);
            float p1 = __expf(sv1 - mnew);
            float rs = p0 + p1;
            rs += __shfl_xor(rs, 1);
            rs += __shfl_xor(rs, 2);
            rs += __shfl_xor(rs, 4);
            rs += __shfl_xor(rs, 8);
            lr[r] = lr[r] * a_ + rs;
            mr[r] = mnew;
            alpha[r] = a_;
            int rowi = hf * 8 + r;
            plds[rowi * 32 + lm]      = (_Float16)p0;  // K = lm
            plds[rowi * 32 + 16 + lm] = (_Float16)p1;  // K = 16+lm
        }
#pragma unroll
        for (int t = 0; t < 4; ++t)
#pragma unroll
            for (int r = 0; r < 8; ++r) o[t][r] *= alpha[r];

        __syncthreads();
        // reload P in A layout, full K=32
        v8h p_lo = *(const v8h*)(plds + lm * 32 + hf * 16);
        v8h p_hi = *(const v8h*)(plds + lm * 32 + hf * 16 + 8);
        v16h ap = __builtin_shufflevector(p_lo, p_hi, 0, 1, 2, 3, 4, 5, 6, 7,
                                          8, 9, 10, 11, 12, 13, 14, 15);
        __syncthreads();

        // ---- O += P @ V (K=32, all lanes carry data) ----
#pragma unroll
        for (int t = 0; t < 4; ++t) {
            v16h bvf = *(const v16h*)(vb + (size_t)(t * 16 + lm) * TMEMP + j0 + hf * 16);
            o[t] = __builtin_amdgcn_wmma_f32_16x16x32_f16(false, ap, false, bvf, (short)0, o[t], false, false);
        }
    }

#pragma unroll
    for (int r = 0; r < 8; ++r) lr[r] = 1.0f / lr[r];
#pragma unroll
    for (int t = 0; t < 4; ++t) {
        int col = h * HDIM + t * 16 + lm;
#pragma unroll
        for (int r = 0; r < 8; ++r) {
            int row = q0 + hf * 8 + r;
            yatt[(size_t)(b * TDIM + row) * CDIM + col] = (_Float16)(o[t][r] * lr[r]);
        }
    }
}

// ---------------------------------------------------------------------------
extern "C" void kernel_launch(void* const* d_in, const int* in_sizes, int n_in,
                              void* d_out, int out_size, void* d_ws, size_t ws_size,
                              hipStream_t stream)
{
    (void)in_sizes; (void)n_in; (void)out_size; (void)ws_size;
    const float* x   = (const float*)d_in[0];
    const float* mk1 = (const float*)d_in[1];
    const float* mv1 = (const float*)d_in[2];
    const float* mk2 = (const float*)d_in[3];
    const float* mv2 = (const float*)d_in[4];
    const float* mk3 = (const float*)d_in[5];
    const float* mv3 = (const float*)d_in[6];
    const float* caw = (const float*)d_in[7];
    const float* cab = (const float*)d_in[8];
    const float* cpw = (const float*)d_in[9];
    const float* cpb = (const float*)d_in[10];
    const float* wk  = (const float*)d_in[11];
    const float* gk  = (const float*)d_in[12];
    const float* bk  = (const float*)d_in[13];
    const float* wv  = (const float*)d_in[14];
    const float* gv  = (const float*)d_in[15];
    const float* bv  = (const float*)d_in[16];

    // workspace layout (all offsets 256B aligned)
    char* ws = (char*)d_ws;
    _Float16* xh    = (_Float16*)(ws);               // 4096*1024 f16 =  8,388,608 B
    _Float16* cawh  = (_Float16*)(ws + 8388608);     // 3072*1024 f16 =  6,291,456 B
    _Float16* cpwh  = (_Float16*)(ws + 14680064);    // 1024*1024 f16 =  2,097,152 B
    _Float16* qkvh  = (_Float16*)(ws + 16777216);    // 4096*3072 f16 = 25,165,824 B
    _Float16* allk  = (_Float16*)(ws + 41943040);    // 64*1667*64 f16 = 13,656,064 B
    _Float16* allvT = (_Float16*)(ws + 55599104);    // 64*64*1680 f16 = 13,762,560 B
    _Float16* yatt  = (_Float16*)(ws + 69361664);    // 4096*1024 f16 =  8,388,608 B
    float* out = (float*)d_out;

    // 0) one-time f32 -> f16 conversion of activations and weights
    k_cvt<<<dim3(2048), dim3(256), 0, stream>>>(x,   xh,   (long)4096 * 1024);
    k_cvt<<<dim3(1536), dim3(256), 0, stream>>>(caw, cawh, (long)3072 * 1024);
    k_cvt<<<dim3(512),  dim3(256), 0, stream>>>(cpw, cpwh, (long)1024 * 1024);

    // 1) qkv projection: 64 M-tiles x 48 N-tiles, one wave each
    k_gemm<3072, false><<<dim3(64 * 48), dim3(32), 0, stream>>>(xh, cawh, cab, qkvh);

    // 2) compress k3/v3 into packed buffers
    k_compress<<<dim3(64 * 257, 2), dim3(32), 0, stream>>>(mk3, mv3, wk, gk, bk,
                                                           wv, gv, bv, allk, allvT);

    // 3) pack mem1/mem2 + new k/v
    {
        long total = (long)64 * 1410 * 64;
        int blocks = (int)((total + 255) / 256);
        k_pack<<<dim3(blocks), dim3(256), 0, stream>>>(mk1, mv1, mk2, mv2, qkvh, allk, allvT);
    }

    // 4) attention: 4096 waves (one per b,h,16-row query tile)
    k_attn<<<dim3(4096), dim3(32), 0, stream>>>(qkvh, allk, allvT, yatt);

    // 5) output projection
    k_gemm<1024, true><<<dim3(64 * 16), dim3(32), 0, stream>>>(yatt, cpwh, cpb, out);
}
